// MixtralMoE_5634997092485
// MI455X (gfx1250) — compile-verified, matrix operand order checked by reading, not run on hardware
//
#include <hip/hip_runtime.h>
#include <hip/hip_bf16.h>
#include <stdint.h>

// Problem constants (match reference)
#define T_TOK 2048
#define H_DIM 2048
#define N_EXP 8
#define I_DIM 4096

// Tiling
#define BT    16      // tokens per block (one WMMA M tile)
#define BN    128     // N columns per pass (8 waves x 16)
#define BK    32      // K per WMMA step (bf16 16x16x32)
#define NTHR  256     // 8 waves
#define A_LD  2056    // (2048 + 8) halfs, padded row stride -> conflict-free b128
#define S_LD  40      // (32 + 8) halfs, staging tile row stride

typedef __attribute__((ext_vector_type(16))) __bf16 v16bf;
typedef __attribute__((ext_vector_type(8)))  __bf16 v8bf;
typedef __attribute__((ext_vector_type(8)))  float  v8f;

union PkBf4 { __bf16 b[4]; unsigned long long u; };
union FragU { v16bf v; v8bf h[2]; };

// ---------------------------------------------------------------------------
// Kernel 0: zero output + routing counters (harness poisons d_out/d_ws)
// ---------------------------------------------------------------------------
__global__ void moe_zero_kernel(float* __restrict__ out, int n, int* __restrict__ cnt) {
    int i = blockIdx.x * blockDim.x + threadIdx.x;
    if (i < 16) cnt[i] = 0;
    for (int j = i; j < n; j += gridDim.x * blockDim.x) out[j] = 0.0f;
}

// ---------------------------------------------------------------------------
// Kernel 1: router. logits -> exp -> top2 -> renormalize -> compacted lists.
// ---------------------------------------------------------------------------
__global__ __launch_bounds__(NTHR) void moe_router_kernel(
    const float* __restrict__ hidden, const float* __restrict__ gate,
    int* __restrict__ cnt, int* __restrict__ ptok, float* __restrict__ pw) {
    __shared__ float gsh[N_EXP * H_DIM];   // 64 KB
    const int tid = threadIdx.x;
    for (int i = tid * 4; i < N_EXP * H_DIM; i += NTHR * 4)
        *(float4*)&gsh[i] = *(const float4*)&gate[i];
    __syncthreads();

    const int t = blockIdx.x * NTHR + tid;
    float acc[N_EXP];
#pragma unroll
    for (int e = 0; e < N_EXP; ++e) acc[e] = 0.0f;

    const float4* hrow = (const float4*)(hidden + (size_t)t * H_DIM);
    for (int k4 = 0; k4 < H_DIM / 4; ++k4) {
        float4 x = hrow[k4];
#pragma unroll
        for (int e = 0; e < N_EXP; ++e) {
            float4 g = *(const float4*)&gsh[e * H_DIM + k4 * 4];
            acc[e] += x.x * g.x + x.y * g.y + x.z * g.z + x.w * g.w;
        }
    }
    float m = acc[0];
#pragma unroll
    for (int e = 1; e < N_EXP; ++e) m = fmaxf(m, acc[e]);
    float p[N_EXP];
#pragma unroll
    for (int e = 0; e < N_EXP; ++e) p[e] = __expf(acc[e] - m);

    int i1 = 0;
#pragma unroll
    for (int e = 1; e < N_EXP; ++e) if (p[e] > p[i1]) i1 = e;
    int i2 = (i1 == 0) ? 1 : 0;
#pragma unroll
    for (int e = 0; e < N_EXP; ++e) if (e != i1 && p[e] > p[i2]) i2 = e;

    const float inv = __builtin_amdgcn_rcpf(p[i1] + p[i2]);  // renorm (softmax denom cancels)
    const float w1 = p[i1] * inv, w2 = p[i2] * inv;

    int s1 = atomicAdd(&cnt[i1], 1);
    ptok[i1 * T_TOK + s1] = t;  pw[i1 * T_TOK + s1] = w1;
    int s2 = atomicAdd(&cnt[i2], 1);
    ptok[i2 * T_TOK + s2] = t;  pw[i2 * T_TOK + s2] = w2;
}

// ---------------------------------------------------------------------------
// Fragment helpers (ISA 7.12.2 layouts, wave32).
// A (16xK bf16): lanes 0-15 row M=lane, K in {k0..k0+7, k0+16..k0+23};
//                lanes 16-31 same rows, K in {k0+8.., k0+24..}.
// B mirrored: N = lane&15 across lanes, same K split by lane half.
// ---------------------------------------------------------------------------
__device__ __forceinline__ v16bf load_frag_a(const __bf16* buf, int ld, int k0, int lane) {
    const int r = lane & 15;
    const int koff = (lane < 16) ? 0 : 8;
    const __bf16* p = buf + r * ld + k0 + koff;
    FragU u;
    u.h[0] = *(const v8bf*)p;          // ds_load_b128
    u.h[1] = *(const v8bf*)(p + 16);   // ds_load_b128
    return u.v;
}

__device__ __forceinline__ v16bf load_frag_b(const __bf16* stg, int ns, int lane) {
    const int c = lane & 15;
    const int koff = (lane < 16) ? 0 : 8;
    const __bf16* p = stg + (ns * 16 + c) * S_LD + koff;
    FragU u;
    u.h[0] = *(const v8bf*)p;
    u.h[1] = *(const v8bf*)(p + 16);
    return u.v;
}

// Stage a [BN x BK] fp32 weight tile -> bf16 LDS (fragment-friendly layout).
// 256 threads: thread -> (row = tid>>1, 16-col half = tid&1), 4x float4 each.
__device__ __forceinline__ void stage_tile(__bf16* __restrict__ dst,
                                           const float* __restrict__ src,
                                           int ld, int kbase, int tid) {
    const int r = tid >> 1;
    const int hh = tid & 1;
    const float* p = src + (size_t)r * ld + kbase + hh * 16;
    __builtin_prefetch(p + BK, 0, 1);   // global_prefetch_b8: tile after next
#pragma unroll
    for (int j = 0; j < 4; ++j) {
        float4 x = *(const float4*)(p + j * 4);
        PkBf4 pk;
        pk.b[0] = (__bf16)x.x; pk.b[1] = (__bf16)x.y;
        pk.b[2] = (__bf16)x.z; pk.b[3] = (__bf16)x.w;
        *(unsigned long long*)&dst[r * S_LD + hh * 16 + j * 4] = pk.u;
    }
}

// ---------------------------------------------------------------------------
// Kernel 2: fused expert MLP. One block = (16 routed tokens) x (1 expert).
// GEMM1 (x @ W13^T) + SwiGLU into LDS strip, GEMM2 (h @ W2^T) -> weighted
// atomic combine. Two I-halves; double-buffered weight staging (one barrier
// per K-step; global loads of tile ks+1 overlap ds_load+wmma of tile ks).
// ---------------------------------------------------------------------------
__global__ __launch_bounds__(NTHR, 1) void moe_expert_kernel(
    const float* __restrict__ hidden, const float* __restrict__ w13,
    const float* __restrict__ w2, float* __restrict__ out,
    const int* __restrict__ cnt, const int* __restrict__ ptok,
    const float* __restrict__ pw) {

    __shared__ __align__(16) __bf16 Abuf[BT * A_LD];        // 65,792 B token acts
    __shared__ __align__(16) __bf16 Hbuf[BT * A_LD];        // 65,792 B swiglu half-strip
    __shared__ __align__(16) __bf16 stgG[2][BN * S_LD];     // 2 x 10,240 B (gate / gemm2)
    __shared__ __align__(16) __bf16 stgU[2][BN * S_LD];     // 2 x 10,240 B (up)
    __shared__ int   tokS[BT];
    __shared__ float wS[BT];

    const int e = blockIdx.y;
    const int ntok = cnt[e];
    const int base = blockIdx.x * BT;
    if (base >= ntok) return;

    const int tid  = threadIdx.x;
    const int wave = tid >> 5;
    const int lane = tid & 31;

    if (tid < BT) {
        int slot = base + tid;
        if (slot < ntok) { tokS[tid] = ptok[e * T_TOK + slot]; wS[tid] = pw[e * T_TOK + slot]; }
        else             { tokS[tid] = ptok[e * T_TOK + base]; wS[tid] = 0.0f; }
    }
    __syncthreads();

    // Gather 16 token rows, fp32 -> bf16, into padded LDS A buffer.
    for (int idx = tid; idx < BT * (H_DIM / 4); idx += NTHR) {
        int r  = idx / (H_DIM / 4);
        int c4 = idx % (H_DIM / 4);
        float4 x = *(const float4*)&hidden[(size_t)tokS[r] * H_DIM + c4 * 4];
        PkBf4 pk;
        pk.b[0] = (__bf16)x.x; pk.b[1] = (__bf16)x.y;
        pk.b[2] = (__bf16)x.z; pk.b[3] = (__bf16)x.w;
        *(unsigned long long*)&Abuf[r * A_LD + c4 * 4] = pk.u;
    }

    const float* w13e = w13 + (size_t)e * (2 * I_DIM) * H_DIM;
    const float* w2e  = w2  + (size_t)e * H_DIM * I_DIM;
    const int mAdd = (lane < 16) ? 0 : 8;

    for (int half = 0; half < 2; ++half) {
        const int iBase = half * (I_DIM / 2);

        // ---- GEMM1 + SwiGLU: fill Hbuf[:, 0..2048) for this I-half ----
        for (int p = 0; p < (I_DIM / 2) / BN; ++p) {        // 16 passes
            v8f accG = {}; v8f accU = {};
            const int rowG = iBase + p * BN;
            const int rowU = I_DIM + rowG;
            const float* srcG = w13e + (size_t)rowG * H_DIM;
            const float* srcU = w13e + (size_t)rowU * H_DIM;
            constexpr int KS1 = H_DIM / BK;                 // 64
            stage_tile(stgG[0], srcG, H_DIM, 0, tid);       // prologue: tile 0
            stage_tile(stgU[0], srcU, H_DIM, 0, tid);
            for (int ks = 0; ks < KS1; ++ks) {
                __syncthreads();                            // tile ks&1 ready
                if (ks + 1 < KS1) {                         // stage next tile
                    stage_tile(stgG[(ks + 1) & 1], srcG, H_DIM, (ks + 1) * BK, tid);
                    stage_tile(stgU[(ks + 1) & 1], srcU, H_DIM, (ks + 1) * BK, tid);
                }
                v16bf a  = load_frag_a(Abuf, A_LD, ks * BK, lane);
                v16bf bg = load_frag_b(stgG[ks & 1], wave, lane);
                v16bf bu = load_frag_b(stgU[ks & 1], wave, lane);
                accG = __builtin_amdgcn_wmma_f32_16x16x32_bf16(
                    false, a, false, bg, (short)0, accG, false, false);
                accU = __builtin_amdgcn_wmma_f32_16x16x32_bf16(
                    false, a, false, bu, (short)0, accU, false, false);
            }
            const int nCol = p * BN + wave * 16 + (lane & 15);
#pragma unroll
            for (int v = 0; v < 8; ++v) {
                float g = accG[v], u = accU[v];
                float sig = __builtin_amdgcn_rcpf(1.0f + __expf(-g));  // v_rcp_f32
                Hbuf[(v + mAdd) * A_LD + nCol] = (__bf16)(g * sig * u);
            }
        }
        __syncthreads();   // Hbuf complete before GEMM2 reads

        // ---- GEMM2 partial: out += (h_half @ W2[:, iBase..]^T) * weight ----
        for (int p = 0; p < H_DIM / BN; ++p) {              // 16 passes
            v8f accO = {};
            const float* srcW = w2e + (size_t)(p * BN) * I_DIM;
            constexpr int KS2 = (I_DIM / 2) / BK;           // 64
            stage_tile(stgG[0], srcW, I_DIM, iBase, tid);   // prologue: tile 0
            for (int ks = 0; ks < KS2; ++ks) {
                __syncthreads();
                if (ks + 1 < KS2)
                    stage_tile(stgG[(ks + 1) & 1], srcW, I_DIM,
                               iBase + (ks + 1) * BK, tid);
                v16bf a = load_frag_a(Hbuf, A_LD, ks * BK, lane);
                v16bf b = load_frag_b(stgG[ks & 1], wave, lane);
                accO = __builtin_amdgcn_wmma_f32_16x16x32_bf16(
                    false, a, false, b, (short)0, accO, false, false);
            }
            const int nCol = p * BN + wave * 16 + (lane & 15);
#pragma unroll
            for (int v = 0; v < 8; ++v) {
                int m = v + mAdd;
                atomicAdd(&out[(size_t)tokS[m] * H_DIM + nCol], accO[v] * wS[m]);
            }
            __syncthreads();   // stg reads done before next pass prologue
        }
        __syncthreads();       // Hbuf reads done before next half rewrites it
    }
}

// ---------------------------------------------------------------------------
extern "C" void kernel_launch(void* const* d_in, const int* in_sizes, int n_in,
                              void* d_out, int out_size, void* d_ws, size_t ws_size,
                              hipStream_t stream) {
    (void)in_sizes; (void)n_in; (void)out_size; (void)ws_size;
    const float* hidden = (const float*)d_in[0];   // [T, H]
    const float* gate   = (const float*)d_in[1];   // [E, H]
    const float* w13    = (const float*)d_in[2];   // [E, 2I, H]
    const float* w2     = (const float*)d_in[3];   // [E, H, I]
    float* out = (float*)d_out;                    // [T, H]

    // Workspace layout: counters, per-expert token lists, per-expert weights.
    int*   cnt  = (int*)d_ws;                      // 16 ints
    int*   ptok = cnt + 16;                        // E*T ints
    float* pw   = (float*)(ptok + N_EXP * T_TOK);  // E*T floats

    moe_zero_kernel<<<dim3(2048), dim3(NTHR), 0, stream>>>(out, T_TOK * H_DIM, cnt);
    moe_router_kernel<<<dim3(T_TOK / NTHR), dim3(NTHR), 0, stream>>>(
        hidden, gate, cnt, ptok, pw);
    moe_expert_kernel<<<dim3(T_TOK / BT, N_EXP), dim3(NTHR), 0, stream>>>(
        hidden, w13, w2, out, cnt, ptok, pw);
}
